// MultiHeadSelfAttention_1846835937544
// MI455X (gfx1250) — matrix operand order repaired and checked by reading, not compiled, not run on hardware
//
#include <hip/hip_runtime.h>
#include <hip/hip_bf16.h>
#include <cstdint>
#include <cstddef>

// ---------------------------------------------------------------------------
// Types matching the gfx1250 WMMA builtin signatures
// ---------------------------------------------------------------------------
typedef __attribute__((ext_vector_type(16))) __bf16 v16bf;
typedef __attribute__((ext_vector_type(8)))  __bf16 v8bf;
typedef __attribute__((ext_vector_type(8)))  float  v8f;

union Frag16 { v16bf f; v8bf h[2]; };

#define BATCH 2
#define SEQ   4096
#define EMBED 768
#define NQKV  2304
#define HEADS 12
#define HDIM  64

__device__ __forceinline__ __bf16 f2bf(float x) {
  uint32_t u = __builtin_bit_cast(uint32_t, x);
  uint32_t r = u + 0x7FFFu + ((u >> 16) & 1u);   // round-to-nearest-even
  uint16_t hv = (uint16_t)(r >> 16);
  return __builtin_bit_cast(__bf16, hv);
}

__device__ __forceinline__ v8f zero8() {
  v8f z = {0.f,0.f,0.f,0.f,0.f,0.f,0.f,0.f};
  return z;
}

__device__ __forceinline__ v8f wmma_bf16(const Frag16& a, const Frag16& b, v8f c) {
  return __builtin_amdgcn_wmma_f32_16x16x32_bf16(
      /*neg_a=*/false, a.f, /*neg_b=*/false, b.f,
      /*c_mod=*/(short)0, c, /*reuse_a=*/false, /*reuse_b=*/false);
}

// ---------------------------------------------------------------------------
// Kernel 1a: fp32 -> bf16 conversion (grid-stride)
// ---------------------------------------------------------------------------
__global__ void f32_to_bf16_kernel(const float* __restrict__ in,
                                   __bf16* __restrict__ out, int n) {
  int i = blockIdx.x * blockDim.x + threadIdx.x;
  int stride = gridDim.x * blockDim.x;
  for (; i < n; i += stride) out[i] = f2bf(in[i]);
}

// ---------------------------------------------------------------------------
// Kernel 1b: fp32 [K][N] -> bf16 [N][K] transpose-convert (weights are tiny)
// ---------------------------------------------------------------------------
__global__ void transpose_to_bf16_kernel(const float* __restrict__ in,  // [K][N]
                                         __bf16* __restrict__ out,      // [N][K]
                                         int K, int N) {
  int idx = blockIdx.x * blockDim.x + threadIdx.x;
  int total = K * N;
  int stride = gridDim.x * blockDim.x;
  for (; idx < total; idx += stride) {
    int k = idx / N, n = idx - k * N;      // coalesced read over n
    out[(size_t)n * K + k] = f2bf(in[idx]);
  }
}

// ---------------------------------------------------------------------------
// Kernel 2: QKV projection GEMM  [8192,768] x [768,2304]^T-staged + bias
//   Wt is pre-transposed [2304, 768] so A and B tiles stage identically
//   with b128 copies. Double-buffered LDS, one barrier per k-step.
//   Writes Q (scaled 1/sqrt(D)) and K as [B,H,S,D] bf16,
//   V transposed as [B,H,D,S] bf16.
// ---------------------------------------------------------------------------
__global__ __launch_bounds__(256) void qkv_gemm_kernel(
    const __bf16* __restrict__ X,     // [8192, 768]
    const __bf16* __restrict__ Wt,    // [2304, 768]  (pre-transposed)
    const float*  __restrict__ bias,  // [2304]
    __bf16* __restrict__ Qb,
    __bf16* __restrict__ Kb,
    __bf16* __restrict__ Vt)
{
  __shared__ __align__(16) __bf16 As[2][128 * 32];   // [m][k]
  __shared__ __align__(16) __bf16 Bs[2][128 * 32];   // [n][k]

  const int tid  = threadIdx.x;
  const int wave = tid >> 5, lane = tid & 31;
  const int half = lane >> 4, l16 = lane & 15;
  const int wm = wave >> 2, wn = wave & 3;
  const int m0 = blockIdx.y * 128;
  const int n0 = blockIdx.x * 128;

  const int srow = tid >> 1, sseg = tid & 1;         // staging: 2 thr/row, 16 bf16
  const __bf16* aSrc = X  + (size_t)(m0 + srow) * EMBED + sseg * 16;
  const __bf16* bSrc = Wt + (size_t)(n0 + srow) * EMBED + sseg * 16;

  v8f acc[4][2];
  #pragma unroll
  for (int i = 0; i < 4; i++)
    #pragma unroll
    for (int j = 0; j < 2; j++) acc[i][j] = zero8();

  auto stage = [&](int k0, int buf) {
    const v8bf* sa = (const v8bf*)(aSrc + k0);
    v8bf a0 = sa[0], a1 = sa[1];
    *(v8bf*)(&As[buf][srow * 32 + sseg * 16])     = a0;
    *(v8bf*)(&As[buf][srow * 32 + sseg * 16 + 8]) = a1;
    const v8bf* sb = (const v8bf*)(bSrc + k0);
    v8bf b0 = sb[0], b1 = sb[1];
    *(v8bf*)(&Bs[buf][srow * 32 + sseg * 16])     = b0;
    *(v8bf*)(&Bs[buf][srow * 32 + sseg * 16 + 8]) = b1;
  };

  constexpr int NIT = EMBED / 32;                    // 24
  stage(0, 0);
  __syncthreads();

  for (int it = 0; it < NIT; ++it) {
    const int buf = it & 1;
    if (it + 1 < NIT) stage((it + 1) * 32, buf ^ 1);
    if (it + 2 < NIT) {                              // global_prefetch_b8
      __builtin_prefetch(aSrc + (it + 2) * 32, 0, 1);
      __builtin_prefetch(bSrc + (it + 2) * 32, 0, 1);
    }

    Frag16 a[4], b[2];
    #pragma unroll
    for (int mf = 0; mf < 4; mf++) {
      const __bf16* base = &As[buf][(wm * 64 + mf * 16 + l16) * 32];
      a[mf].h[0] = *(const v8bf*)(base + half * 8);
      a[mf].h[1] = *(const v8bf*)(base + 16 + half * 8);
    }
    #pragma unroll
    for (int nf = 0; nf < 2; nf++) {
      const __bf16* base = &Bs[buf][(wn * 32 + nf * 16 + l16) * 32];
      b[nf].h[0] = *(const v8bf*)(base + half * 8);
      b[nf].h[1] = *(const v8bf*)(base + 16 + half * 8);
    }
    #pragma unroll
    for (int mf = 0; mf < 4; mf++)
      #pragma unroll
      for (int nf = 0; nf < 2; nf++)
        acc[mf][nf] = wmma_bf16(a[mf], b[nf], acc[mf][nf]);

    __syncthreads();
  }

  // Epilogue: bias add (hoisted per lane-column), scatter into Q/K/V layouts
  #pragma unroll
  for (int nf = 0; nf < 2; nf++) {
    const int n = n0 + wn * 32 + nf * 16 + l16;
    const float bn = bias[n];
    const int which = n / EMBED, r = n % EMBED;
    const int hh = r / HDIM, d = r % HDIM;
    #pragma unroll
    for (int mf = 0; mf < 4; mf++) {
      #pragma unroll
      for (int e = 0; e < 8; e++) {
        int m = m0 + wm * 64 + mf * 16 + e + 8 * half;
        int bi = m >> 12, s = m & (SEQ - 1);
        size_t bh = (size_t)(bi * HEADS + hh);
        float v = acc[mf][nf][e] + bn;
        if (which == 0) {
          Qb[(bh * SEQ + s) * HDIM + d] = f2bf(v * 0.125f);   // 1/sqrt(64)
        } else if (which == 1) {
          Kb[(bh * SEQ + s) * HDIM + d] = f2bf(v);
        } else {
          Vt[(bh * HDIM + d) * SEQ + s] = f2bf(v);
        }
      }
    }
  }
}

// ---------------------------------------------------------------------------
// Kernel 3: causal flash attention.
//   Block = (b,h, 128-query tile); wave w owns rows [q0+16w, q0+16w+15].
//   Streams 32-key blocks; QK^T and P*V on v_wmma_f32_16x16x32_bf16.
//   K/V/Q fragments are contiguous global b128 loads (layouts arranged so).
// ---------------------------------------------------------------------------
__global__ __launch_bounds__(256) void flash_attn_kernel(
    const __bf16* __restrict__ Qb,   // [B,H,S,D], pre-scaled
    const __bf16* __restrict__ Kb,   // [B,H,S,D]
    const __bf16* __restrict__ Vt,   // [B,H,D,S]
    __bf16* __restrict__ Att)        // [B,S,E]
{
  __shared__ __align__(16) __bf16 Pl[8 * 16 * 32];  // per-wave P staging

  const int tid  = threadIdx.x;
  const int wave = tid >> 5, lane = tid & 31;
  const int half = lane >> 4, l16 = lane & 15;
  const int q0 = blockIdx.x * 128;
  const int hh = blockIdx.y, bi = blockIdx.z;
  const size_t bh = (size_t)(bi * HEADS + hh);
  const int qrow = q0 + wave * 16;

  // Q A-fragments for both 32-wide D steps, straight from global
  Frag16 qf[2];
  #pragma unroll
  for (int ds = 0; ds < 2; ds++) {
    const __bf16* base = Qb + (bh * SEQ + qrow + l16) * HDIM + ds * 32;
    qf[ds].h[0] = *(const v8bf*)(base + half * 8);
    qf[ds].h[1] = *(const v8bf*)(base + 16 + half * 8);
  }

  float m_e[8], l_e[8];
  #pragma unroll
  for (int e = 0; e < 8; e++) { m_e[e] = -3.0e38f; l_e[e] = 0.f; }
  v8f o[4];
  #pragma unroll
  for (int nf = 0; nf < 4; nf++) o[nf] = zero8();

  __bf16* pw = Pl + wave * (16 * 32);
  const int kend = q0 + 128;

  for (int kb = 0; kb < kend; kb += 32) {
    if (kb <= qrow + 15) {              // wave-uniform: EXEC stays all-1s
      if (kb + 32 < kend) {             // prefetch next key block
        __builtin_prefetch(Kb + (bh * SEQ + kb + 32 + l16) * HDIM, 0, 1);
        __builtin_prefetch(Vt + (bh * HDIM + l16) * SEQ + kb + 32, 0, 1);
      }
      // ---- scores: S[16x32] = Q[16x64] * K^T ----
      v8f s0 = zero8(), s1 = zero8();
      #pragma unroll
      for (int ds = 0; ds < 2; ds++) {
        Frag16 b0, b1;
        const __bf16* kbase = Kb + (bh * SEQ + kb + l16) * HDIM + ds * 32;
        b0.h[0] = *(const v8bf*)(kbase + half * 8);
        b0.h[1] = *(const v8bf*)(kbase + 16 + half * 8);
        const __bf16* kbase1 = kbase + 16 * HDIM;
        b1.h[0] = *(const v8bf*)(kbase1 + half * 8);
        b1.h[1] = *(const v8bf*)(kbase1 + 16 + half * 8);
        s0 = wmma_bf16(qf[ds], b0, s0);
        s1 = wmma_bf16(qf[ds], b1, s1);
      }
      // ---- causal mask + online softmax (row = e + 8*half in C layout) ----
      float alpha_e[8], p0v[8], p1v[8];
      #pragma unroll
      for (int e = 0; e < 8; e++) {
        int row = qrow + e + 8 * half;
        float v0 = ((kb + l16)      <= row) ? s0[e] : -3.0e38f;
        float v1 = ((kb + 16 + l16) <= row) ? s1[e] : -3.0e38f;
        float mx = fmaxf(v0, v1);
        #pragma unroll
        for (int off = 1; off < 16; off <<= 1)
          mx = fmaxf(mx, __shfl_xor(mx, off, 32));
        float mnew = fmaxf(m_e[e], mx);
        float al = __expf(m_e[e] - mnew);
        float p0 = __expf(v0 - mnew);
        float p1 = __expf(v1 - mnew);
        float rs = p0 + p1;
        #pragma unroll
        for (int off = 1; off < 16; off <<= 1)
          rs += __shfl_xor(rs, off, 32);
        l_e[e] = l_e[e] * al + rs;
        m_e[e] = mnew;
        alpha_e[e] = al;
        p0v[e] = p0; p1v[e] = p1;
      }
      // ---- C-layout -> A-layout via wave-private LDS ----
      #pragma unroll
      for (int e = 0; e < 8; e++) {
        int rl = e + 8 * half;
        pw[rl * 32 + l16]      = f2bf(p0v[e]);
        pw[rl * 32 + 16 + l16] = f2bf(p1v[e]);
      }
      Frag16 pf;
      pf.h[0] = *(const v8bf*)(pw + l16 * 32 + half * 8);
      pf.h[1] = *(const v8bf*)(pw + l16 * 32 + 16 + half * 8);
      // ---- rescale O, then O += P * V ----
      #pragma unroll
      for (int nf = 0; nf < 4; nf++)
        #pragma unroll
        for (int e = 0; e < 8; e++)
          o[nf][e] *= alpha_e[e];
      #pragma unroll
      for (int nf = 0; nf < 4; nf++) {
        Frag16 vf;
        const __bf16* vbase = Vt + (bh * HDIM + nf * 16 + l16) * SEQ + kb;
        vf.h[0] = *(const v8bf*)(vbase + half * 8);
        vf.h[1] = *(const v8bf*)(vbase + 16 + half * 8);
        o[nf] = wmma_bf16(pf, vf, o[nf]);
      }
    }
  }

  // normalize and write attended [B,S,E]
  #pragma unroll
  for (int nf = 0; nf < 4; nf++) {
    #pragma unroll
    for (int e = 0; e < 8; e++) {
      int row = qrow + e + 8 * half;
      int d = nf * 16 + l16;
      float val = o[nf][e] / l_e[e];
      Att[((size_t)bi * SEQ + row) * EMBED + hh * HDIM + d] = f2bf(val);
    }
  }
}

// ---------------------------------------------------------------------------
// Kernel 4: output projection  [8192,768] x [768,768] + bias -> fp32 d_out
//   Same double-buffered structure; Wt pre-transposed [N][K].
// ---------------------------------------------------------------------------
__global__ __launch_bounds__(256) void out_proj_kernel(
    const __bf16* __restrict__ A,     // attended, [8192, 768]
    const __bf16* __restrict__ Wt,    // [768, 768] (pre-transposed)
    const float*  __restrict__ bias,  // [768]
    float* __restrict__ Out)          // [8192, 768]
{
  __shared__ __align__(16) __bf16 As[2][128 * 32];
  __shared__ __align__(16) __bf16 Bs[2][128 * 32];

  const int tid  = threadIdx.x;
  const int wave = tid >> 5, lane = tid & 31;
  const int half = lane >> 4, l16 = lane & 15;
  const int wm = wave >> 2, wn = wave & 3;
  const int m0 = blockIdx.y * 128;
  const int n0 = blockIdx.x * 128;

  const int srow = tid >> 1, sseg = tid & 1;
  const __bf16* aSrc = A  + (size_t)(m0 + srow) * EMBED + sseg * 16;
  const __bf16* bSrc = Wt + (size_t)(n0 + srow) * EMBED + sseg * 16;

  v8f acc[4][2];
  #pragma unroll
  for (int i = 0; i < 4; i++)
    #pragma unroll
    for (int j = 0; j < 2; j++) acc[i][j] = zero8();

  auto stage = [&](int k0, int buf) {
    const v8bf* sa = (const v8bf*)(aSrc + k0);
    v8bf a0 = sa[0], a1 = sa[1];
    *(v8bf*)(&As[buf][srow * 32 + sseg * 16])     = a0;
    *(v8bf*)(&As[buf][srow * 32 + sseg * 16 + 8]) = a1;
    const v8bf* sb = (const v8bf*)(bSrc + k0);
    v8bf b0 = sb[0], b1 = sb[1];
    *(v8bf*)(&Bs[buf][srow * 32 + sseg * 16])     = b0;
    *(v8bf*)(&Bs[buf][srow * 32 + sseg * 16 + 8]) = b1;
  };

  constexpr int NIT = EMBED / 32;
  stage(0, 0);
  __syncthreads();

  for (int it = 0; it < NIT; ++it) {
    const int buf = it & 1;
    if (it + 1 < NIT) stage((it + 1) * 32, buf ^ 1);
    if (it + 2 < NIT) {
      __builtin_prefetch(aSrc + (it + 2) * 32, 0, 1);
      __builtin_prefetch(bSrc + (it + 2) * 32, 0, 1);
    }

    Frag16 a[4], b[2];
    #pragma unroll
    for (int mf = 0; mf < 4; mf++) {
      const __bf16* base = &As[buf][(wm * 64 + mf * 16 + l16) * 32];
      a[mf].h[0] = *(const v8bf*)(base + half * 8);
      a[mf].h[1] = *(const v8bf*)(base + 16 + half * 8);
    }
    #pragma unroll
    for (int nf = 0; nf < 2; nf++) {
      const __bf16* base = &Bs[buf][(wn * 32 + nf * 16 + l16) * 32];
      b[nf].h[0] = *(const v8bf*)(base + half * 8);
      b[nf].h[1] = *(const v8bf*)(base + 16 + half * 8);
    }
    #pragma unroll
    for (int mf = 0; mf < 4; mf++)
      #pragma unroll
      for (int nf = 0; nf < 2; nf++)
        acc[mf][nf] = wmma_bf16(a[mf], b[nf], acc[mf][nf]);

    __syncthreads();
  }

  #pragma unroll
  for (int nf = 0; nf < 2; nf++) {
    const int n = n0 + wn * 32 + nf * 16 + l16;
    const float bn = bias[n];
    #pragma unroll
    for (int mf = 0; mf < 4; mf++) {
      #pragma unroll
      for (int e = 0; e < 8; e++) {
        int m = m0 + wm * 64 + mf * 16 + e + 8 * half;
        Out[(size_t)m * EMBED + n] = acc[mf][nf][e] + bn;
      }
    }
  }
}

// ---------------------------------------------------------------------------
// Launcher
// ---------------------------------------------------------------------------
extern "C" void kernel_launch(void* const* d_in, const int* in_sizes, int n_in,
                              void* d_out, int out_size, void* d_ws, size_t ws_size,
                              hipStream_t stream) {
  const float* x     = (const float*)d_in[0];
  const float* w_qkv = (const float*)d_in[1];
  const float* b_qkv = (const float*)d_in[2];
  const float* w_out = (const float*)d_in[3];
  const float* b_out = (const float*)d_in[4];
  float* out = (float*)d_out;

  const size_t nx    = (size_t)BATCH * SEQ * EMBED;   // 6,291,456
  const size_t nwqkv = (size_t)EMBED * NQKV;          // 1,769,472
  const size_t nwout = (size_t)EMBED * EMBED;         //   589,824

  char* ws = (char*)d_ws;
  auto bump = [&](size_t bytes) -> char* {
    char* p = ws;
    ws += (bytes + 255) & ~(size_t)255;
    return p;
  };
  __bf16* xb    = (__bf16*)bump(nx * 2);
  __bf16* wqkvt = (__bf16*)bump(nwqkv * 2);   // [2304, 768]
  __bf16* woutt = (__bf16*)bump(nwout * 2);   // [768, 768]
  __bf16* Qb    = (__bf16*)bump(nx * 2);
  __bf16* Kb    = (__bf16*)bump(nx * 2);
  __bf16* Vt    = (__bf16*)bump(nx * 2);
  __bf16* Att   = (__bf16*)bump(nx * 2);

  f32_to_bf16_kernel<<<2048, 256, 0, stream>>>(x, xb, (int)nx);
  transpose_to_bf16_kernel<<<1024, 256, 0, stream>>>(w_qkv, wqkvt, EMBED, NQKV);
  transpose_to_bf16_kernel<<<512, 256, 0, stream>>>(w_out, woutt, EMBED, EMBED);

  dim3 g1(NQKV / 128, (BATCH * SEQ) / 128);   // 18 x 64
  qkv_gemm_kernel<<<g1, 256, 0, stream>>>(xb, wqkvt, b_qkv, Qb, Kb, Vt);

  dim3 g2(SEQ / 128, HEADS, BATCH);           // 32 x 12 x 2
  flash_attn_kernel<<<g2, 256, 0, stream>>>(Qb, Kb, Vt, Att);

  dim3 g3(EMBED / 128, (BATCH * SEQ) / 128);  // 6 x 64
  out_proj_kernel<<<g3, 256, 0, stream>>>(Att, woutt, b_out, out);
}
